// NeuralCache_8211977470228
// MI455X (gfx1250) — matrix operand order repaired and checked by reading, compiled, not run on hardware
//
#include <hip/hip_runtime.h>
#include <hip/hip_fp16.h>

typedef __attribute__((ext_vector_type(16))) _Float16 v16h;
typedef __attribute__((ext_vector_type(8)))  float    v8f;

#define WAVES_PER_BLOCK 8
#define EMBD 16
#define HID  64
#define KPAD 32

__device__ __forceinline__ void lds_fence() {
  // per-wave LDS ordering: wait all outstanding DS ops, block compiler reordering
  asm volatile("s_wait_dscnt 0" ::: "memory");
}

// Load a 16x32 f16 A-fragment from LDS laid out row-major [16][strideH].
// Layout: lane m (m<16) holds K = koff+0..15 (elems 0..15);
// lane m+16 holds K = koff+16..31.
__device__ __forceinline__ v16h load_a16(const _Float16* base, int strideH, int koff, int lane) {
  const int row = lane & 15;
  const int k0  = koff + ((lane & 16) ? 16 : 0);
  const _Float16* r0 = base + row * strideH + k0;
  v16h a;
  ((uint4*)&a)[0] = *(const uint4*)(r0);        // K = k0..k0+7
  ((uint4*)&a)[1] = *(const uint4*)(r0 + 8);    // K = k0+8..k0+15
  return a;
}

__global__ __launch_bounds__(256)
void neural_cache_kernel(const float* __restrict__ pos,
                         const float* __restrict__ normal,
                         const float* __restrict__ emb,
                         const float* __restrict__ W1, const float* __restrict__ b1,
                         const float* __restrict__ W2, const float* __restrict__ b2,
                         const float* __restrict__ W3, const float* __restrict__ b3,
                         float* __restrict__ out, int nTiles)
{
  __shared__ _Float16 s_feat[WAVES_PER_BLOCK][16][KPAD];   // 1 KB / wave
  __shared__ _Float16 s_h[WAVES_PER_BLOCK][16][HID];       // 2 KB / wave
  __shared__ float    s_stage[WAVES_PER_BLOCK][96];        // pos/normal/out staging

  const int  lane = threadIdx.x & 31;
  const int  w    = threadIdx.x >> 5;
  const int  p    = lane & 15;
  const bool up   = lane >= 16;
  const int  kb   = up ? 16 : 0;      // B-fragment K base for this lane half
  const int  rofs = up ? 8 : 0;       // C/D row offset for this lane half

  // ---------------- weight fragments, resident in VGPRs ----------------
  // B layout (16-bit, KxN=32x16): lanes 0-15 hold K=0..15 of column N=lane,
  // lanes 16-31 hold K=16..31 of column N=lane-16; element j <-> K=kb+j.
  v16h w1f[4], w2f[4][2], w3f[2];
  float b1v[4], b2v[4], b3v;
  #pragma unroll
  for (int t = 0; t < 4; ++t) {
    const int col = 16 * t + p;
    #pragma unroll
    for (int j = 0; j < 16; ++j) {
      const int k = kb + j;
      w1f[t][j] = (k < EMBD + 3) ? (_Float16)W1[k * HID + col] : (_Float16)0.0f;
    }
    #pragma unroll
    for (int h = 0; h < 2; ++h)
      #pragma unroll
      for (int j = 0; j < 16; ++j)
        w2f[t][h][j] = (_Float16)W2[(32 * h + kb + j) * HID + col];
    b1v[t] = b1[col];
    b2v[t] = b2[col];
  }
  #pragma unroll
  for (int h = 0; h < 2; ++h)
    #pragma unroll
    for (int j = 0; j < 16; ++j)
      w3f[h][j] = (p < 3) ? (_Float16)W3[(32 * h + kb + j) * 3 + p] : (_Float16)0.0f;
  b3v = (p < 3) ? b3[p] : 0.0f;

  const int waveId     = blockIdx.x * WAVES_PER_BLOCK + w;
  const int waveStride = gridDim.x * WAVES_PER_BLOCK;

  for (int tile = waveId; tile < nTiles; tile += waveStride) {
    const long long base = (long long)tile * 16;

    // prefetch next tile's streaming inputs
    if (tile + waveStride < nTiles) {
      const long long nb = (long long)(tile + waveStride) * 16 * 3;
      __builtin_prefetch(pos + nb + (lane << 1), 0, 3);
      __builtin_prefetch(normal + nb + (lane << 1), 0, 3);
    }

    // ---- stage pos (48 f32) + normal (48 f32), coalesced b128 ----
    if (lane < 12)
      ((float4*)&s_stage[w][0])[lane] = ((const float4*)(pos + base * 3))[lane];
    else if (lane >= 16 && lane < 28)
      ((float4*)&s_stage[w][48])[lane - 16] = ((const float4*)(normal + base * 3))[lane - 16];
    lds_fence();

    // ---- hash + gather -> f16 feature tile [16 x 32] (19 valid K) ----
    {
      const float* st = &s_stage[w][0];
      const float px = st[3 * p + 0], py = st[3 * p + 1], pz = st[3 * p + 2];
      const float nx = st[48 + 3 * p + 0], ny = st[48 + 3 * p + 1], nz = st[48 + 3 * p + 2];
      // low 12 bits of the int64 hash equal the low 12 bits of 32-bit math
      const int sx = (int)(px * 10.0f), sy = (int)(py * 10.0f), sz = (int)(pz * 10.0f);
      const unsigned hsh = (unsigned)sx * 73856093u ^ (unsigned)sy * 19349663u
                         ^ (unsigned)sz * 83492791u;
      const int idx = (int)(hsh & 4095u);
      const float* erow = emb + idx * EMBD;
      if (!up) {
        #pragma unroll
        for (int j = 0; j < 8; ++j) s_feat[w][p][j] = (_Float16)erow[j];
        s_feat[w][p][16] = (_Float16)nx;
        s_feat[w][p][17] = (_Float16)ny;
        s_feat[w][p][18] = (_Float16)nz;
        #pragma unroll
        for (int j = 19; j < 24; ++j) s_feat[w][p][j] = (_Float16)0.0f;
      } else {
        #pragma unroll
        for (int j = 8; j < 16; ++j) s_feat[w][p][j] = (_Float16)erow[j];
        #pragma unroll
        for (int j = 24; j < 32; ++j) s_feat[w][p][j] = (_Float16)0.0f;
      }
    }
    lds_fence();

    // ---- layer 1: [16x32] x [32x64] (4 N-tiles) ----
    v8f acc[4];
    {
      const v16h a1 = load_a16(&s_feat[w][0][0], KPAD, 0, lane);
      #pragma unroll
      for (int t = 0; t < 4; ++t) {
        v8f c;
        #pragma unroll
        for (int r = 0; r < 8; ++r) c[r] = b1v[t];
        acc[t] = __builtin_amdgcn_wmma_f32_16x16x32_f16(false, a1, false, w1f[t],
                                                        (short)0, c, false, false);
      }
    }
    #pragma unroll
    for (int t = 0; t < 4; ++t)
      #pragma unroll
      for (int r = 0; r < 8; ++r) {
        float v = acc[t][r];
        v = v > 0.0f ? v : 0.0f;
        s_h[w][r + rofs][16 * t + p] = (_Float16)v;   // relu -> f16 activation tile
      }
    lds_fence();

    // ---- layer 2: [16x64] x [64x64], K split 2x32 ----
    {
      const v16h a20 = load_a16(&s_h[w][0][0], HID, 0, lane);
      const v16h a21 = load_a16(&s_h[w][0][0], HID, 32, lane);
      lds_fence();  // activation reads done before overwrite below
      #pragma unroll
      for (int t = 0; t < 4; ++t) {
        v8f c;
        #pragma unroll
        for (int r = 0; r < 8; ++r) c[r] = b2v[t];
        c = __builtin_amdgcn_wmma_f32_16x16x32_f16(false, a20, false, w2f[t][0],
                                                   (short)0, c, false, false);
        acc[t] = __builtin_amdgcn_wmma_f32_16x16x32_f16(false, a21, false, w2f[t][1],
                                                        (short)0, c, false, false);
      }
    }
    #pragma unroll
    for (int t = 0; t < 4; ++t)
      #pragma unroll
      for (int r = 0; r < 8; ++r) {
        float v = acc[t][r];
        v = v > 0.0f ? v : 0.0f;
        s_h[w][r + rofs][16 * t + p] = (_Float16)v;
      }
    lds_fence();

    // ---- layer 3: [16x64] x [64x3 padded to 16] ----
    v8f c3;
    {
      const v16h a30 = load_a16(&s_h[w][0][0], HID, 0, lane);
      const v16h a31 = load_a16(&s_h[w][0][0], HID, 32, lane);
      #pragma unroll
      for (int r = 0; r < 8; ++r) c3[r] = b3v;
      c3 = __builtin_amdgcn_wmma_f32_16x16x32_f16(false, a30, false, w3f[0],
                                                  (short)0, c3, false, false);
      c3 = __builtin_amdgcn_wmma_f32_16x16x32_f16(false, a31, false, w3f[1],
                                                  (short)0, c3, false, false);
    }

    // ---- sigmoid, stage [16x3] f32, coalesced b128 store ----
    if (p < 3) {
      #pragma unroll
      for (int r = 0; r < 8; ++r) {
        const float v = c3[r];
        s_stage[w][(r + rofs) * 3 + p] = 1.0f / (1.0f + __expf(-v));
      }
    }
    lds_fence();
    if (lane < 12)
      ((float4*)(out + base * 3))[lane] = ((const float4*)&s_stage[w][0])[lane];
    lds_fence();  // staging buffer reused next iteration
  }
}

extern "C" void kernel_launch(void* const* d_in, const int* in_sizes, int n_in,
                              void* d_out, int out_size, void* d_ws, size_t ws_size,
                              hipStream_t stream) {
  const float* pos    = (const float*)d_in[0];
  const float* normal = (const float*)d_in[1];
  const float* emb    = (const float*)d_in[2];
  const float* W1     = (const float*)d_in[3];
  const float* b1     = (const float*)d_in[4];
  const float* W2     = (const float*)d_in[5];
  const float* b2     = (const float*)d_in[6];
  const float* W3     = (const float*)d_in[7];
  const float* b3     = (const float*)d_in[8];
  float* out = (float*)d_out;

  const int N      = in_sizes[0] / 3;   // 4,194,304 (divisible by 16)
  const int nTiles = N / 16;
  int blocks = (nTiles + WAVES_PER_BLOCK - 1) / WAVES_PER_BLOCK;
  if (blocks > 4096) blocks = 4096;     // ~32K waves; each loops grid-stride
  if (blocks < 1) blocks = 1;

  neural_cache_kernel<<<blocks, 256, 0, stream>>>(pos, normal, emb,
                                                  W1, b1, W2, b2, W3, b3,
                                                  out, nTiles);
}
